// QuantizedLinearINT4_26731876451111
// MI455X (gfx1250) — compile-verified, matrix operand order checked by reading, not run on hardware
//
#include <hip/hip_runtime.h>
#include <hip/hip_bf16.h>
#include <stdint.h>

typedef __attribute__((ext_vector_type(16))) _Float16 v16h;
typedef __attribute__((ext_vector_type(8)))  _Float16 v8h;
typedef __attribute__((ext_vector_type(4)))  _Float16 v4h;
typedef __attribute__((ext_vector_type(8)))  float    v8f;

#define IN_DIM  4096
#define OUT_DIM 4096
#define BM 128
#define BN 128
#define BK 32
#define NK (IN_DIM / BK)          // 128 K-tiles
#define NBUF 4                    // async pipeline depth (circular LDS buffers)
#define BUF_HALFS (BM * BK)       // 4096 halfs per buffer
#define BUF_BYTES (BUF_HALFS * 2) // 8192 B per buffer

// LDS tiles are stored in WMMA *fragment order* (per CDNA5 ISA 16-bit A/B layout):
//   a 16(row)x32(K) f16 subtile -> frag[lane][j], lane = half*16 + row,
//   where for element k:  half = (k>>3)&1,  j = ((k&16)>>1) + (k&7)
// so each lane's v16h fragment is 32 contiguous bytes in LDS, and any
// 8-aligned group of k maps to 8 contiguous halfs (one b128 transfer).

// ---------------------------------------------------------------------------
// Pre-pass 1: x f32 -> f16 (memory-bound)
// ---------------------------------------------------------------------------
__global__ __launch_bounds__(256)
void cvt_x_f16(const float* __restrict__ x, _Float16* __restrict__ xh)
{
    const size_t base = ((size_t)blockIdx.x * 256 + threadIdx.x) * 8;
    const float4 f0 = *(const float4*)(x + base);
    const float4 f1 = *(const float4*)(x + base + 4);
    v8h h;
    h[0] = (_Float16)f0.x; h[1] = (_Float16)f0.y;
    h[2] = (_Float16)f0.z; h[3] = (_Float16)f0.w;
    h[4] = (_Float16)f1.x; h[5] = (_Float16)f1.y;
    h[6] = (_Float16)f1.z; h[7] = (_Float16)f1.w;
    *(v8h*)(xh + base) = h;
}

// ---------------------------------------------------------------------------
// Pre-pass 2: int4 weights -> dequantized f16 [OUT, IN] row-major
// ---------------------------------------------------------------------------
__global__ __launch_bounds__(256)
void dequant_w_f16(const uint32_t* __restrict__ wpk,   // one byte-value per int32
                   const float* __restrict__ scale,
                   const float* __restrict__ zp,
                   _Float16* __restrict__ wh)
{
    const size_t nib = ((size_t)blockIdx.x * 256 + threadIdx.x) * 8;
    const int n = (int)(nib >> 12);                    // / IN_DIM
    const uint4 w = *(const uint4*)(wpk + (nib >> 1));
    const float s   = scale[n];
    const float nzs = -zp[n] * s;
    uint32_t q[4] = {w.x, w.y, w.z, w.w};
    v8h h;
#pragma unroll
    for (int qq = 0; qq < 4; ++qq) {
        h[2*qq+0] = (_Float16)fmaf((float)( q[qq]       & 15u), s, nzs);
        h[2*qq+1] = (_Float16)fmaf((float)((q[qq] >> 4) & 15u), s, nzs);
    }
    *(v8h*)(wh + nib) = h;
}

// ---------------------------------------------------------------------------
// Fast GEMM: 4-deep async-to-LDS pipeline, pure WMMA hot loop.
// Per K-step per thread: 4x global_load_async_to_lds_b128 (swizzled dest),
// 12x ds_load_b128 fragment reads per wave, 8 WMMAs per wave, 1 barrier.
// ~3 K-tiles of global-load latency stay hidden behind WMMA work.
// ---------------------------------------------------------------------------
__global__ __launch_bounds__(256, 1)
void gemm_wmma_f16_async(const _Float16* __restrict__ xh,
                         const _Float16* __restrict__ wh,
                         const float* __restrict__ bias,
                         float* __restrict__ out)
{
    __shared__ __align__(32) _Float16 sA[NBUF][BUF_HALFS];   // 4 x 8 KB
    __shared__ __align__(32) _Float16 sB[NBUF][BUF_HALFS];   // 4 x 8 KB

    const int tid   = threadIdx.x;
    const int lane  = tid & 31;
    const int wid   = tid >> 5;
    const int waveM = wid & 3;         // 4 waves along M, 32 rows each
    const int waveN = wid >> 2;        // 2 waves along N, 64 cols each

    const int bM = blockIdx.y * BM;
    const int bN = blockIdx.x * BN;

    // Each tile (128 rows x 32 halfs) = 512 b128 chunks; 2 per thread.
    // chunk c: row = c>>2, k0 = (c&3)*8  (8-aligned -> contiguous b128 in LDS)
    uint64_t aSrc[2], bSrc[2];
    uint32_t aDstB[2], bDstB[2];       // LDS byte address of buffer-0 slot
#pragma unroll
    for (int t = 0; t < 2; ++t) {
        int c    = tid + t * 256;
        int r    = c >> 2;
        int k0   = (c & 3) * 8;
        int half = (k0 >> 3) & 1;
        int j0   = (k0 >> 4) * 8;
        int off  = ((r >> 4) * 512) + (half * 16 + (r & 15)) * 16 + j0;
        aSrc[t]  = (uint64_t)(uintptr_t)(xh + (size_t)(bM + r) * IN_DIM + k0);
        bSrc[t]  = (uint64_t)(uintptr_t)(wh + (size_t)(bN + r) * IN_DIM + k0);
        aDstB[t] = (uint32_t)(uintptr_t)&sA[0][off];   // flat addr low 32b = LDS offset
        bDstB[t] = (uint32_t)(uintptr_t)&sB[0][off];
    }

    uint32_t bsel = 0;   // byte offset of the buffer the NEXT async issue fills

    auto issue_async = [&]() {
#pragma unroll
        for (int t = 0; t < 2; ++t) {
            uint32_t ad = aDstB[t] + bsel;
            uint32_t bd = bDstB[t] + bsel;
            asm volatile("global_load_async_to_lds_b128 %0, %1, off"
                         :: "v"(ad), "v"(aSrc[t]) : "memory");
            asm volatile("global_load_async_to_lds_b128 %0, %1, off"
                         :: "v"(bd), "v"(bSrc[t]) : "memory");
            aSrc[t] += BK * sizeof(_Float16);   // advance 64 B to next K-tile
            bSrc[t] += BK * sizeof(_Float16);
        }
        bsel = (bsel + (uint32_t)BUF_BYTES) & (uint32_t)(NBUF * BUF_BYTES - 1);
    };

    v8f acc[2][4];
#pragma unroll
    for (int i = 0; i < 2; ++i)
#pragma unroll
        for (int j = 0; j < 4; ++j)
#pragma unroll
            for (int r = 0; r < 8; ++r) acc[i][j][r] = 0.0f;

    // ---- prologue: fill buffers 0..2 (3 tiles in flight) ----
    issue_async();                                   // tile 0 -> buf 0
    issue_async();                                   // tile 1 -> buf 1
    issue_async();                                   // tile 2 -> buf 2
    asm volatile("s_wait_asynccnt 0x8" ::: "memory");   // tile 0 complete
    __syncthreads();

    // ---- main K loop: single barrier per iteration ----
    for (int kt = 0; kt < NK; ++kt) {
        const int cur = kt & (NBUF - 1);
        if (kt + 3 < NK) issue_async();              // tile kt+3 -> buf (kt+3)&3
                                                     // (== buf (kt-1)&3, consumed last iter)
        v16h a[2], b[4];
#pragma unroll
        for (int i = 0; i < 2; ++i)
            a[i] = *(const v16h*)&sA[cur][((waveM * 2 + i) * 512) + lane * 16];
#pragma unroll
        for (int j = 0; j < 4; ++j)
            b[j] = *(const v16h*)&sB[cur][((waveN * 4 + j) * 512) + lane * 16];

#pragma unroll
        for (int i = 0; i < 2; ++i)
#pragma unroll
            for (int j = 0; j < 4; ++j)
                acc[i][j] = __builtin_amdgcn_wmma_f32_16x16x32_f16(
                    false, a[i], false, b[j], (short)0, acc[i][j], false, false);

        // Ensure tile kt+1 has landed (async loads complete in order: the 8
        // newest ops belong to tiles kt+2/kt+3). In the tail no issues are
        // outstanding beyond kt+1, so wait for everything.
        if (kt + 3 < NK) asm volatile("s_wait_asynccnt 0x8" ::: "memory");
        else             asm volatile("s_wait_asynccnt 0x0" ::: "memory");
        __syncthreads();                 // publish tile kt+1; all done reading buf[cur]
    }

    // ---- epilogue: C/D layout (m = r + 8*(lane>=16), n = lane&15) ----
    const int half = lane >> 4;
    const int nloc = lane & 15;
#pragma unroll
    for (int i = 0; i < 2; ++i) {
#pragma unroll
        for (int j = 0; j < 4; ++j) {
            const int mBase = bM + waveM * 32 + i * 16 + half * 8;
            const int nG    = bN + waveN * 64 + j * 16 + nloc;
            const float bv  = bias[nG];
#pragma unroll
            for (int r = 0; r < 8; ++r)
                out[(size_t)(mBase + r) * OUT_DIM + nG] = acc[i][j][r] + bv;
        }
    }
}

// ---------------------------------------------------------------------------
// Fallback: fully fused kernel (dequant + convert in the K loop), used when
// the workspace is too small for the pre-converted operands.
// ---------------------------------------------------------------------------
__global__ __launch_bounds__(256, 1)
void q4_gemm_wmma_fused(const float* __restrict__ x,
                        const uint32_t* __restrict__ wpk,
                        const float* __restrict__ scale,
                        const float* __restrict__ zp,
                        const float* __restrict__ bias,
                        float* __restrict__ out)
{
    __shared__ __align__(32) _Float16 sA[2][BM * BK];
    __shared__ __align__(32) _Float16 sB[2][BN * BK];

    const int tid   = threadIdx.x;
    const int lane  = tid & 31;
    const int wid   = tid >> 5;
    const int waveM = wid & 3;
    const int waveN = wid >> 2;

    const int bM = blockIdx.y * BM;
    const int bN = blockIdx.x * BN;

    int aLds[4];
#pragma unroll
    for (int t = 0; t < 4; ++t) {
        int c  = tid + t * 256;
        int m  = c >> 3;
        int k0 = (c & 7) * 4;
        int half = (k0 >> 3) & 1;
        int j0   = ((k0 & 16) >> 1) + (k0 & 7);
        aLds[t]  = ((m >> 4) * 512) + (half * 16 + (m & 15)) * 16 + j0;
    }
    int bLds[2];
#pragma unroll
    for (int t = 0; t < 2; ++t) {
        int c  = tid + t * 256;
        int n  = c >> 2;
        int k0 = (c & 3) * 8;
        int half = (k0 >> 3) & 1;
        int j0   = (k0 >> 4) * 8;
        bLds[t]  = ((n >> 4) * 512) + (half * 16 + (n & 15)) * 16 + j0;
    }

    v4h aH[4];
    v8h bH[2];

    auto fetch_convert = [&](int kt) {
#pragma unroll
        for (int t = 0; t < 4; ++t) {
            int c  = tid + t * 256;
            int m  = c >> 3;
            int k0 = (c & 7) * 4;
            const float4 f = *(const float4*)(x + (size_t)(bM + m) * IN_DIM
                                                + (size_t)kt * BK + k0);
            v4h h;
            h[0] = (_Float16)f.x; h[1] = (_Float16)f.y;
            h[2] = (_Float16)f.z; h[3] = (_Float16)f.w;
            aH[t] = h;
        }
#pragma unroll
        for (int t = 0; t < 2; ++t) {
            int c  = tid + t * 256;
            int n  = c >> 2;
            int k0 = (c & 3) * 8;
            int nG = bN + n;
            size_t e0 = ((size_t)nG * IN_DIM + (size_t)kt * BK + (size_t)k0) >> 1;
            const uint4 w = *(const uint4*)(wpk + e0);
            const float s   = scale[nG];
            const float nzs = -zp[nG] * s;
            uint32_t q[4] = {w.x, w.y, w.z, w.w};
            v8h h;
#pragma unroll
            for (int qq = 0; qq < 4; ++qq) {
                h[2*qq+0] = (_Float16)fmaf((float)( q[qq]       & 15u), s, nzs);
                h[2*qq+1] = (_Float16)fmaf((float)((q[qq] >> 4) & 15u), s, nzs);
            }
            bH[t] = h;
        }
    };
    auto store_tiles = [&](int buf) {
#pragma unroll
        for (int t = 0; t < 4; ++t) *(v4h*)&sA[buf][aLds[t]] = aH[t];
#pragma unroll
        for (int t = 0; t < 2; ++t) *(v8h*)&sB[buf][bLds[t]] = bH[t];
    };

    v8f acc[2][4];
#pragma unroll
    for (int i = 0; i < 2; ++i)
#pragma unroll
        for (int j = 0; j < 4; ++j)
#pragma unroll
            for (int r = 0; r < 8; ++r) acc[i][j][r] = 0.0f;

    fetch_convert(0);
    store_tiles(0);
    __syncthreads();

    for (int kt = 0; kt < NK; ++kt) {
        const int cur = kt & 1;
        if (kt + 1 < NK) fetch_convert(kt + 1);

        v16h a[2], b[4];
#pragma unroll
        for (int i = 0; i < 2; ++i)
            a[i] = *(const v16h*)&sA[cur][((waveM * 2 + i) * 512) + lane * 16];
#pragma unroll
        for (int j = 0; j < 4; ++j)
            b[j] = *(const v16h*)&sB[cur][((waveN * 4 + j) * 512) + lane * 16];

#pragma unroll
        for (int i = 0; i < 2; ++i)
#pragma unroll
            for (int j = 0; j < 4; ++j)
                acc[i][j] = __builtin_amdgcn_wmma_f32_16x16x32_f16(
                    false, a[i], false, b[j], (short)0, acc[i][j], false, false);

        __syncthreads();
        if (kt + 1 < NK) {
            store_tiles(1 - cur);
            __syncthreads();
        }
    }

    const int half = lane >> 4;
    const int nloc = lane & 15;
#pragma unroll
    for (int i = 0; i < 2; ++i) {
#pragma unroll
        for (int j = 0; j < 4; ++j) {
            const int mBase = bM + waveM * 32 + i * 16 + half * 8;
            const int nG    = bN + waveN * 64 + j * 16 + nloc;
            const float bv  = bias[nG];
#pragma unroll
            for (int r = 0; r < 8; ++r)
                out[(size_t)(mBase + r) * OUT_DIM + nG] = acc[i][j][r] + bv;
        }
    }
}

extern "C" void kernel_launch(void* const* d_in, const int* in_sizes, int n_in,
                              void* d_out, int out_size, void* d_ws, size_t ws_size,
                              hipStream_t stream) {
    const float*    xin  = (const float*)d_in[0];
    const uint32_t* wpk  = (const uint32_t*)d_in[1];
    const float*    scl  = (const float*)d_in[2];
    const float*    zpt  = (const float*)d_in[3];
    const float*    bias = (const float*)d_in[4];
    float*          out  = (float*)d_out;

    const int M = in_sizes[0] / IN_DIM;          // 8192
    dim3 grid(OUT_DIM / BN, M / BM);             // (32, 64)

    const size_t xh_bytes = (size_t)M * IN_DIM * sizeof(_Float16);
    const size_t wh_bytes = (size_t)OUT_DIM * IN_DIM * sizeof(_Float16);

    if (ws_size >= xh_bytes + wh_bytes) {
        _Float16* xh = (_Float16*)d_ws;
        _Float16* wh = (_Float16*)((char*)d_ws + xh_bytes);

        const int xBlocks = (int)(((size_t)M * IN_DIM / 8) / 256);        // 16384
        const int wBlocks = (int)(((size_t)OUT_DIM * IN_DIM / 8) / 256);  // 8192
        cvt_x_f16    <<<xBlocks, 256, 0, stream>>>(xin, xh);
        dequant_w_f16<<<wBlocks, 256, 0, stream>>>(wpk, scl, zpt, wh);
        gemm_wmma_f16_async<<<grid, 256, 0, stream>>>(xh, wh, bias, out);
    } else {
        q4_gemm_wmma_fused<<<grid, 256, 0, stream>>>(xin, wpk, scl, zpt, bias, out);
    }
}
